// get_model_75350906241417
// MI455X (gfx1250) — matrix operand order repaired and checked by reading, not compiled, not run on hardware
//
#include <hip/hip_runtime.h>
#include <hip/hip_bf16.h>

// ---------------------------------------------------------------------------
// DGCNN forward for MI455X (gfx1250, wave32, WMMA + TDM).
//  - all GEMMs on v_wmma_f32_16x16x32_f16 with f16 pre-converted weights
//  - neighbor gather via TENSOR_LOAD_TO_LDS gather-mode descriptors (TDM),
//    LDS row padding done by the DMA engine; s_wait_tensorcnt sync
//  - center-term WMMA factored out of the k loop (1 tile instead of 3)
//  - gemm5: one A fragment amortized over 8 ptiles (8 accumulators/wave)
//  - max-over-k fused into the WMMA accumulators via lane shuffles
//  - BN+LeakyReLU folded after max (monotone-commute)
// ---------------------------------------------------------------------------

typedef _Float16 f16;
typedef __attribute__((ext_vector_type(16))) _Float16 v16h;
typedef __attribute__((ext_vector_type(8)))  float    v8f;
typedef __attribute__((ext_vector_type(4)))  unsigned int u32x4;
typedef __attribute__((ext_vector_type(8)))  int          i32x8;
typedef __attribute__((ext_vector_type(4)))  int          i32x4;

#define NPTS    1024
#define BATCH   8
#define P_TOTAL (BATCH * NPTS)
#define KNN     40
#define KPAD    48          // 3 WMMA N-tiles of 16
#define EMB     1024
#define EPS     1e-5f

#if defined(__HIP_DEVICE_COMPILE__) && __has_builtin(__builtin_amdgcn_tensor_load_to_lds)
#define USE_TDM 1
#else
#define USE_TDM 0
#endif

// -------------------------------------------------- weights f32 -> padded f16
__global__ void prepw_kernel(const float* __restrict__ src, f16* __restrict__ dst,
                             int O, int Csrc, int Cpad) {
    int t = blockIdx.x * blockDim.x + threadIdx.x;
    if (t >= O * Cpad) return;
    int o = t / Cpad, kk = t % Cpad;
    dst[t] = (kk < Csrc) ? (f16)src[o * Csrc + kk] : (f16)0.f;
}

// -------------------------------------------------- prep: x -> xT0 f16 + xx
__global__ void prep0_kernel(const float* __restrict__ x,
                             f16* __restrict__ xT0, float* __restrict__ xx) {
    int p = blockIdx.x * blockDim.x + threadIdx.x;
    if (p >= P_TOTAL) return;
    int b = p >> 10, n = p & 1023;
    float s = 0.f;
    #pragma unroll
    for (int c = 0; c < 8; ++c) {
        float v = (c < 3) ? x[(b * 3 + c) * NPTS + n] : 0.f;
        xT0[p * 8 + c] = (f16)v;
        s += v * v;
    }
    xx[p] = s;
}

// -------------------------------------------------- per-slice |x|^2
__global__ void xx_kernel(const f16* __restrict__ xin, int stride, int C,
                          float* __restrict__ xx) {
    int p = blockIdx.x * blockDim.x + threadIdx.x;
    if (p >= P_TOTAL) return;
    float s = 0.f;
    for (int c = 0; c < C; ++c) {
        float v = (float)xin[(size_t)p * stride + c];
        s += v * v;
    }
    xx[p] = s;
}

// -------------------------------------------------- kNN: top-40 of 2*inner - |xn|^2 - |xm|^2
template <int C>
__global__ __launch_bounds__(256)
void knn_kernel(const f16* __restrict__ xin, int stride,
                const float* __restrict__ xx, int* __restrict__ idx) {
    int p = blockIdx.x * blockDim.x + threadIdx.x;
    if (p >= P_TOTAL) return;
    int b = p >> 10;
    float best[KNN];
    int   bidx[KNN];
    #pragma unroll
    for (int i = 0; i < KNN; ++i) { best[i] = -3.4e38f; bidx[i] = 0; }

    f16 ctr[C];                                   // center row in registers
    #pragma unroll
    for (int c = 0; c < C; ++c) ctr[c] = xin[(size_t)p * stride + c];

    float xxn = xx[p];
    const f16* rows = xin + (size_t)(b * NPTS) * stride;
    for (int m = 0; m < NPTS; ++m) {
        const f16* row = rows + (size_t)m * stride;
        __builtin_prefetch(row + stride, 0, 0);   // global_prefetch_b8
        float dot = 0.f;
        #pragma unroll
        for (int c = 0; c < C; ++c) dot += (float)ctr[c] * (float)row[c];
        float pd = 2.f * dot - xxn - xx[b * NPTS + m];
        if (pd > best[KNN - 1]) {                 // descending insertion
            int j = KNN - 1;
            while (j > 0 && best[j - 1] < pd) {
                best[j] = best[j - 1]; bidx[j] = bidx[j - 1]; --j;
            }
            best[j] = pd; bidx[j] = m;
        }
    }
    for (int i = 0; i < KNN; ++i) idx[p * KNN + i] = bidx[i];
}

#if USE_TDM
// Issue one TDM gather-mode load: nrows rows (16-bit indices from sidx[rowBase..])
// of cin f16 each, row-padded by +4 DWORDs in LDS. D# bit layout per ISA 8.3-8.7.
__device__ __forceinline__ void tdm_gather(unsigned lds_addr, unsigned long long gaddr,
                                           int cin, int stride_elems, int nrows,
                                           int padIntCode, const int* sidx, int rowBase) {
    u32x4 g0;
    g0.x = 1u | (1u << 31);                                   // count=1, gather_mode, 16b idx
    g0.y = (unsigned)__builtin_amdgcn_readfirstlane((int)lds_addr);
    g0.z = (unsigned)__builtin_amdgcn_readfirstlane((int)(gaddr & 0xffffffffu));
    g0.w = ((unsigned)__builtin_amdgcn_readfirstlane((int)((gaddr >> 32) & 0x01ffffffu)))
         | (2u << 30);                                        // type=2 (image)
    i32x8 g1;
    g1[0] = (1 << 16) | (1 << 20) | (padIntCode << 22) | (3 << 25); // 2B elems, pad 4DW/row
    g1[1] = (cin & 0xffff) << 16;                             // tensor_dim0 lo16
    g1[2] = (NPTS << 16);                                     // tensor_dim0 hi | tensor_dim1 lo
    g1[3] = (cin << 16);                                      // tensor_dim1 hi | tile_dim0
    g1[4] = nrows;                                            // tile_dim1 = #valid indices
    g1[5] = stride_elems;                                     // tensor_dim0_stride lo32
    g1[6] = 0;
    g1[7] = 0;
    i32x4 g2, g3;
    #pragma unroll
    for (int j = 0; j < 4; ++j) {
        int s0 = (2 * j     < nrows) ? sidx[rowBase + 2 * j]     : 0;
        int s1 = (2 * j + 1 < nrows) ? sidx[rowBase + 2 * j + 1] : 0;
        int s2 = (8 + 2 * j     < nrows) ? sidx[rowBase + 8 + 2 * j]     : 0;
        int s3 = (8 + 2 * j + 1 < nrows) ? sidx[rowBase + 8 + 2 * j + 1] : 0;
        g2[j] = __builtin_amdgcn_readfirstlane((s0 & 0xffff) | (s1 << 16));
        g3[j] = __builtin_amdgcn_readfirstlane((s2 & 0xffff) | (s3 << 16));
    }
#if __clang_major__ >= 23
    __builtin_amdgcn_tensor_load_to_lds(g0, g1, g2, g3, (i32x8)0, 0);
#else
    __builtin_amdgcn_tensor_load_to_lds(g0, g1, g2, g3, 0);
#endif
}
#endif

// -------------------------------------------------- EdgeConv via WMMA (+TDM)
// 1 workgroup (4 waves) per point.
// CIN>=32: TDM gathers raw neighbor rows into LDS nbr[k][CIN+8]; -ctr applied
//          in place; center term gets its own 1-tile WMMA accumulator.
// CIN<32 : manual gather into combined featT[k][2C] (layer 1 only).
template <int CIN, int COUT>
__global__ __launch_bounds__(128)
void edgeconv_kernel(const f16* __restrict__ xin, int in_stride,
                     const int* __restrict__ idx,
                     const f16* __restrict__ wp,    // padded [COUT][ROWS] f16
                     const float* __restrict__ bn,  // [4][COUT]
                     f16* __restrict__ xout, int out_stride, int out_off) {
    constexpr int C2     = 2 * CIN;
    constexpr int CHUNKS = (C2 + 31) / 32;
    constexpr int ROWS   = CHUNKS * 32;            // power of two
    __shared__ int sidx[KNN];
    __shared__ f16 sctr[CIN];

    const int p   = blockIdx.x;
    const int b   = p >> 10;
    const int tid = threadIdx.x;

    if (tid < KNN) sidx[tid] = idx[p * KNN + tid];
    for (int c = tid; c < CIN; c += 128) sctr[c] = xin[(size_t)p * in_stride + c];
    __syncthreads();

    const int wave = tid >> 5;
    const int lane = tid & 31;
    const int lrow = lane & 15;
    const int hi   = (lane & 16) ? 1 : 0;

    if constexpr (CIN >= 32 && USE_TDM) {
#if USE_TDM
        __shared__ f16 nbr[KPAD][CIN + 8];         // rows 40..47 unwritten (masked)
        constexpr int PADC = __builtin_ctz(CIN / 2) - 1;   // pad interval = row DWORDs
        constexpr unsigned ROWB = (CIN + 8) * 2;           // LDS row stride bytes
        if (wave == 0) {
            unsigned lds0 = (unsigned)(size_t)&nbr[0][0];
            unsigned long long ga = (unsigned long long)(size_t)(xin + (size_t)b * NPTS * in_stride);
            tdm_gather(lds0,             ga, CIN, in_stride, 16, PADC, sidx, 0);
            tdm_gather(lds0 + 16 * ROWB, ga, CIN, in_stride, 16, PADC, sidx, 16);
            tdm_gather(lds0 + 32 * ROWB, ga, CIN, in_stride,  8, PADC, sidx, 32);
            __builtin_amdgcn_s_wait_tensorcnt(0);
        }
        __syncthreads();
        // apply -ctr in place (diff features)
        for (int e = tid; e < KNN * CIN; e += 128) {
            int k = e / CIN, c = e & (CIN - 1);
            nbr[k][c] = (f16)((float)nbr[k][c] - (float)sctr[c]);
        }
        __syncthreads();

        constexpr int HCH = CHUNKS / 2;            // chunks covering the diff half
        for (int ot = wave; ot < COUT / 16; ot += 4) {
            v8f a0 = {}, a1 = {}, a2 = {}, ac = {};
            const f16* wrow = wp + (size_t)(ot * 16 + lrow) * ROWS;
            for (int ch = 0; ch < HCH; ++ch) {     // Wa x (nbr - ctr), 3 k-tiles
                v16h A, B0, B1, B2;
                #pragma unroll
                for (int e = 0; e < 16; ++e) {
                    A[e] = wrow[ch * 32 + hi * 16 + e];
                    int cc = ch * 32 + hi * 16 + e;
                    B0[e] = nbr[lrow][cc];
                    B1[e] = nbr[16 + lrow][cc];
                    B2[e] = nbr[32 + lrow][cc];
                }
                a0 = __builtin_amdgcn_wmma_f32_16x16x32_f16(false, A, false, B0, (short)0, a0, false, false);
                a1 = __builtin_amdgcn_wmma_f32_16x16x32_f16(false, A, false, B1, (short)0, a1, false, false);
                a2 = __builtin_amdgcn_wmma_f32_16x16x32_f16(false, A, false, B2, (short)0, a2, false, false);
            }
            for (int ch = HCH; ch < CHUNKS; ++ch) { // Wb x ctr: k-independent, 1 tile
                v16h A, Bc;
                #pragma unroll
                for (int e = 0; e < 16; ++e) {
                    A[e]  = wrow[ch * 32 + hi * 16 + e];
                    Bc[e] = sctr[(ch - HCH) * 32 + hi * 16 + e];
                }
                ac = __builtin_amdgcn_wmma_f32_16x16x32_f16(false, A, false, Bc, (short)0, ac, false, false);
            }
            bool t2valid = lrow < (KNN - 32);
            #pragma unroll
            for (int r = 0; r < 8; ++r) {
                float m = fmaxf(a0[r], a1[r]);
                if (t2valid) m = fmaxf(m, a2[r]);
                #pragma unroll
                for (int s = 1; s <= 8; s <<= 1)
                    m = fmaxf(m, __shfl_xor(m, s, 32));
                m += ac[r];                        // center term (uniform over k)
                if (lrow == 0) {
                    int chn = ot * 16 + r + hi * 8;
                    float g  = bn[chn],            be = bn[COUT + chn];
                    float mu = bn[2 * COUT + chn], vv = bn[3 * COUT + chn];
                    float y = (m - mu) * g * rsqrtf(vv + EPS) + be;
                    y = (y >= 0.f) ? y : 0.2f * y;
                    xout[(size_t)p * out_stride + out_off + chn] = (f16)y;
                }
            }
        }
#endif
    } else {
        // manual combined gather (layer 1 / no-TDM fallback)
        __shared__ f16 featT[KPAD][ROWS + 8];
        for (int e = tid; e < KPAD * ROWS; e += 128) {
            int k = e / ROWS, c = e & (ROWS - 1);
            f16 v = (f16)0.f;
            if (k < KNN && c < C2) {
                if (c < CIN) {
                    int nb = sidx[k];
                    float xv = (float)xin[(size_t)(b * NPTS + nb) * in_stride + c];
                    v = (f16)(xv - (float)sctr[c]);
                } else {
                    v = sctr[c - CIN];
                }
            }
            featT[k][c] = v;
        }
        __syncthreads();

        for (int ot = wave; ot < COUT / 16; ot += 4) {
            v8f a0 = {}, a1 = {}, a2 = {};
            const f16* wrow = wp + (size_t)(ot * 16 + lrow) * ROWS;
            for (int ch = 0; ch < CHUNKS; ++ch) {
                v16h A, B0, B1, B2;
                #pragma unroll
                for (int e = 0; e < 16; ++e) {
                    A[e] = wrow[ch * 32 + hi * 16 + e];
                    int kc = ch * 32 + hi * 16 + e;
                    B0[e] = featT[lrow][kc];
                    B1[e] = featT[16 + lrow][kc];
                    B2[e] = featT[32 + lrow][kc];
                }
                a0 = __builtin_amdgcn_wmma_f32_16x16x32_f16(false, A, false, B0, (short)0, a0, false, false);
                a1 = __builtin_amdgcn_wmma_f32_16x16x32_f16(false, A, false, B1, (short)0, a1, false, false);
                a2 = __builtin_amdgcn_wmma_f32_16x16x32_f16(false, A, false, B2, (short)0, a2, false, false);
            }
            bool t2valid = lrow < (KNN - 32);
            #pragma unroll
            for (int r = 0; r < 8; ++r) {
                float m = fmaxf(a0[r], a1[r]);
                if (t2valid) m = fmaxf(m, a2[r]);
                #pragma unroll
                for (int s = 1; s <= 8; s <<= 1)
                    m = fmaxf(m, __shfl_xor(m, s, 32));
                if (lrow == 0) {
                    int chn = ot * 16 + r + hi * 8;
                    float g  = bn[chn],            be = bn[COUT + chn];
                    float mu = bn[2 * COUT + chn], vv = bn[3 * COUT + chn];
                    float y = (m - mu) * g * rsqrtf(vv + EPS) + be;
                    y = (y >= 0.f) ? y : 0.2f * y;
                    xout[(size_t)p * out_stride + out_off + chn] = (f16)y;
                }
            }
        }
    }
}

// -------------------------------------------------- e = lrelu(bn5(W5 @ xc))
// D[1024 x 8192] = W5[1024x512] x xcT^T.
// One wave per (mtile, 8 ptiles): A fragment reused across 8 B tiles.
__global__ __launch_bounds__(128)
void gemm5_kernel(const f16* __restrict__ xcT, const f16* __restrict__ w5p,
                  const float* __restrict__ bn5, f16* __restrict__ ebuf) {
    int wave = threadIdx.x >> 5, lane = threadIdx.x & 31;
    int tile  = blockIdx.x * 4 + wave;   // 0..4095
    int mtile = tile & 63;               // EMB/16
    int pg    = tile >> 6;               // P_TOTAL/(16*8) = 64 groups
    int lrow  = lane & 15;
    int hi    = (lane & 16) ? 1 : 0;

    v8f acc[8];
    #pragma unroll
    for (int j = 0; j < 8; ++j) acc[j] = (v8f){};

    const f16* wrow = w5p + (size_t)(mtile * 16 + lrow) * 512;
    for (int ch = 0; ch < 512 / 32; ++ch) {
        v16h A;
        #pragma unroll
        for (int e = 0; e < 16; ++e)
            A[e] = wrow[ch * 32 + hi * 16 + e];
        #pragma unroll
        for (int j = 0; j < 8; ++j) {
            const f16* brow = xcT + (size_t)(pg * 128 + j * 16 + lrow) * 512 + ch * 32 + hi * 16;
            v16h B;
            #pragma unroll
            for (int e = 0; e < 16; ++e) B[e] = brow[e];
            acc[j] = __builtin_amdgcn_wmma_f32_16x16x32_f16(false, A, false, B, (short)0, acc[j], false, false);
        }
    }
    #pragma unroll
    for (int j = 0; j < 8; ++j) {
        #pragma unroll
        for (int r = 0; r < 8; ++r) {
            int chn  = mtile * 16 + r + hi * 8;
            int pcol = pg * 128 + j * 16 + lrow;
            float g  = bn5[chn],           be = bn5[EMB + chn];
            float mu = bn5[2 * EMB + chn], vv = bn5[3 * EMB + chn];
            float y = (acc[j][r] - mu) * g * rsqrtf(vv + EPS) + be;
            y = (y >= 0.f) ? y : 0.2f * y;
            ebuf[(size_t)chn * P_TOTAL + pcol] = (f16)y;
        }
    }
}

// -------------------------------------------------- global max+mean pool
__global__ void pool_kernel(const f16* __restrict__ ebuf, float* __restrict__ g) {
    int t = blockIdx.x * blockDim.x + threadIdx.x;  // (b, ch)
    if (t >= BATCH * EMB) return;
    int b = t >> 10, chn = t & 1023;
    const f16* row = ebuf + (size_t)chn * P_TOTAL + b * NPTS;
    float mx = -3.4e38f, sm = 0.f;
    for (int n = 0; n < NPTS; ++n) {
        float v = (float)row[n];
        mx = fmaxf(mx, v); sm += v;
    }
    g[b * 2048 + chn]       = mx;
    g[b * 2048 + EMB + chn] = sm * (1.f / NPTS);
}

// -------------------------------------------------- tiny MLP head (B=8)
__global__ void fc1_kernel(const float* __restrict__ g, const float* __restrict__ l1,
                           const float* __restrict__ bn6, float* __restrict__ h1) {
    int t = blockIdx.x * blockDim.x + threadIdx.x;
    if (t >= BATCH * 512) return;
    int b = t >> 9, o = t & 511;
    float s = 0.f;
    for (int i = 0; i < 2048; ++i) s += g[b * 2048 + i] * l1[o * 2048 + i];
    float y = (s - bn6[1024 + o]) * bn6[o] * rsqrtf(bn6[1536 + o] + EPS) + bn6[512 + o];
    h1[b * 512 + o] = (y >= 0.f) ? y : 0.2f * y;
}

__global__ void fc2_kernel(const float* __restrict__ h1, const float* __restrict__ l2w,
                           const float* __restrict__ l2b, const float* __restrict__ bn7,
                           float* __restrict__ h2) {
    int t = blockIdx.x * blockDim.x + threadIdx.x;
    if (t >= BATCH * 256) return;
    int b = t >> 8, o = t & 255;
    float s = l2b[o];
    for (int i = 0; i < 512; ++i) s += h1[b * 512 + i] * l2w[o * 512 + i];
    float y = (s - bn7[512 + o]) * bn7[o] * rsqrtf(bn7[768 + o] + EPS) + bn7[256 + o];
    h2[b * 256 + o] = (y >= 0.f) ? y : 0.2f * y;
}

__global__ __launch_bounds__(64)
void head_kernel(const float* __restrict__ h2, const float* __restrict__ l3w,
                 const float* __restrict__ l3b, float* __restrict__ out) {
    int b = blockIdx.x, t = threadIdx.x;
    __shared__ float lg[40];
    __shared__ float red[64];
    float v = -3.4e38f;
    if (t < 40) {
        float s = l3b[t];
        for (int i = 0; i < 256; ++i) s += h2[b * 256 + i] * l3w[t * 256 + i];
        lg[t] = s;
        out[b * 40 + t] = s;          // logits
        v = s;
    }
    red[t] = v; __syncthreads();
    for (int s2 = 32; s2 > 0; s2 >>= 1) {
        if (t < s2) red[t] = fmaxf(red[t], red[t + s2]);
        __syncthreads();
    }
    float mx = red[0]; __syncthreads();
    red[t] = (t < 40) ? __expf(lg[t] - mx) : 0.f; __syncthreads();
    for (int s2 = 32; s2 > 0; s2 >>= 1) {
        if (t < s2) red[t] += red[t + s2];
        __syncthreads();
    }
    float lse = __logf(red[0]) + mx;
    if (t < 40) out[320 + b * 40 + t] = lg[t] - lse;   // log_softmax
}

// -------------------------------------------------- launcher
extern "C" void kernel_launch(void* const* d_in, const int* in_sizes, int n_in,
                              void* d_out, int out_size, void* d_ws, size_t ws_size,
                              hipStream_t stream) {
    const float* x   = (const float*)d_in[0];
    const float* w1  = (const float*)d_in[1];
    const float* w2  = (const float*)d_in[2];
    const float* w3  = (const float*)d_in[3];
    const float* w4  = (const float*)d_in[4];
    const float* w5  = (const float*)d_in[5];
    const float* l1  = (const float*)d_in[6];
    const float* l2w = (const float*)d_in[7];
    const float* l2b = (const float*)d_in[8];
    const float* l3w = (const float*)d_in[9];
    const float* l3b = (const float*)d_in[10];
    const float* bn1 = (const float*)d_in[11];
    const float* bn2 = (const float*)d_in[12];
    const float* bn3 = (const float*)d_in[13];
    const float* bn4 = (const float*)d_in[14];
    const float* bn5 = (const float*)d_in[15];
    const float* bn6 = (const float*)d_in[16];
    const float* bn7 = (const float*)d_in[17];

    // workspace layout (bytes)
    char* ws = (char*)d_ws;
    f16*   xcT  = (f16*)(ws);                              //  8192*512  f16 = 8 MB
    f16*   xT0  = (f16*)(ws + 8388608);                    //  8192*8    f16
    int*   idx  = (int*)(ws + 8519680);                    //  8192*40   i32
    float* xx   = (float*)(ws + 9830400);                  //  8192      f32
    f16*   ebuf = (f16*)(ws + 9863168);                    //  1024*8192 f16 = 16 MB
    float* g    = (float*)(ws + 26640384);                 //  8*2048    f32
    float* h1   = (float*)(ws + 26705920);                 //  8*512     f32
    float* h2   = (float*)(ws + 26722304);                 //  8*256     f32
    f16*   w1p  = (f16*)(ws + 26730496);                   //  64*32     f16
    f16*   w2p  = (f16*)(ws + 26734592);                   //  64*128    f16
    f16*   w3p  = (f16*)(ws + 26750976);                   //  128*128   f16
    f16*   w4p  = (f16*)(ws + 26783744);                   //  256*256   f16
    f16*   w5p  = (f16*)(ws + 26914816);                   //  1024*512  f16 = 1 MB

    dim3 b256(256);
    int gP = (P_TOTAL + 255) / 256;   // 32 blocks

    // one-time weight conversion (cheap; part of every call for determinism)
    prepw_kernel<<<(64 * 32 + 255) / 256, b256, 0, stream>>>(w1, w1p, 64, 6, 32);
    prepw_kernel<<<(64 * 128 + 255) / 256, b256, 0, stream>>>(w2, w2p, 64, 128, 128);
    prepw_kernel<<<(128 * 128 + 255) / 256, b256, 0, stream>>>(w3, w3p, 128, 128, 128);
    prepw_kernel<<<(256 * 256 + 255) / 256, b256, 0, stream>>>(w4, w4p, 256, 256, 256);
    prepw_kernel<<<(1024 * 512 + 255) / 256, b256, 0, stream>>>(w5, w5p, 1024, 512, 512);

    // layer 1: C=3 -> 64 (manual gather path)
    prep0_kernel<<<gP, b256, 0, stream>>>(x, xT0, xx);
    knn_kernel<3><<<gP, b256, 0, stream>>>(xT0, 8, xx, idx);
    edgeconv_kernel<3, 64><<<P_TOTAL, 128, 0, stream>>>(xT0, 8, idx, w1p, bn1, xcT, 512, 0);

    // layer 2: C=64 -> 64 (input = xcT[:,0:64])
    xx_kernel<<<gP, b256, 0, stream>>>(xcT, 512, 64, xx);
    knn_kernel<64><<<gP, b256, 0, stream>>>(xcT, 512, xx, idx);
    edgeconv_kernel<64, 64><<<P_TOTAL, 128, 0, stream>>>(xcT, 512, idx, w2p, bn2, xcT, 512, 64);

    // layer 3: C=64 -> 128 (input = xcT[:,64:128])
    xx_kernel<<<gP, b256, 0, stream>>>(xcT + 64, 512, 64, xx);
    knn_kernel<64><<<gP, b256, 0, stream>>>(xcT + 64, 512, xx, idx);
    edgeconv_kernel<64, 128><<<P_TOTAL, 128, 0, stream>>>(xcT + 64, 512, idx, w3p, bn3, xcT, 512, 128);

    // layer 4: C=128 -> 256 (input = xcT[:,128:256])
    xx_kernel<<<gP, b256, 0, stream>>>(xcT + 128, 512, 128, xx);
    knn_kernel<128><<<gP, b256, 0, stream>>>(xcT + 128, 512, xx, idx);
    edgeconv_kernel<128, 256><<<P_TOTAL, 128, 0, stream>>>(xcT + 128, 512, idx, w4p, bn4, xcT, 512, 256);

    // conv5 GEMM + BN + lrelu  (64 mtiles x 64 pgroups, 4 waves/block)
    gemm5_kernel<<<(64 * 64) / 4, 128, 0, stream>>>(xcT, w5p, bn5, ebuf);

    // pooling + head
    pool_kernel<<<(BATCH * EMB + 255) / 256, b256, 0, stream>>>(ebuf, g);
    fc1_kernel<<<(BATCH * 512 + 255) / 256, b256, 0, stream>>>(g, l1, bn6, h1);
    fc2_kernel<<<(BATCH * 256 + 255) / 256, b256, 0, stream>>>(h1, l2w, l2b, bn7, h2);
    head_kernel<<<BATCH, 64, 0, stream>>>(h2, l3w, l3b, (float*)d_out);
}
